// OneScaleMultiStepPredictor_2078764172047
// MI455X (gfx1250) — compile-verified, matrix operand order checked by reading, not run on hardware
//
#include <hip/hip_runtime.h>

typedef __attribute__((ext_vector_type(16))) _Float16    v16h;
typedef __attribute__((ext_vector_type(8)))  float       v8f;
typedef __attribute__((ext_vector_type(4)))  unsigned    u32x4;
typedef __attribute__((ext_vector_type(8)))  int         i32x8;
typedef __attribute__((ext_vector_type(4)))  int         i32x4;

union AFragQ { v16h v; u32x4 q[2]; _Float16 h[16]; };
union CFrag  { v8f  v; float f[8]; };

#define HEMPTY 0xFFFFFFFFFFFFFFFFull

#if defined(__HIP_DEVICE_COMPILE__) && __has_builtin(__builtin_amdgcn_tensor_load_to_lds)
#define HAVE_TDM 1
#else
#define HAVE_TDM 0
#endif

static __device__ __forceinline__ unsigned long long lin3(int x, int y, int z) {
  return (((unsigned long long)(unsigned)(x + 8) * 2048ull) +
          (unsigned long long)(unsigned)(y + 8)) * 2048ull +
         (unsigned long long)(unsigned)(z + 8);
}
static __device__ __forceinline__ unsigned long long hmix(unsigned long long k) {
  k ^= k >> 33; k *= 0xff51afd7ed558ccdull;
  k ^= k >> 33; k *= 0xc4ceb9fe1a85ec53ull;
  k ^= k >> 33;
  return k;
}
static __device__ __forceinline__ float requantf(float v) {
  v *= (1.0f / 256.0f);
  return fminf(fmaxf(v, -128.0f), 127.0f);
}

#if HAVE_TDM
// TDM: flat 8KB global->LDS tile copy (data_size=8B, 1-D tile of 1024 elems).
// D# per CDNA5 ISA ch.8: group0 {count=1, lds_addr, global_addr, type=2},
// group1 {mask=0, data_size=3, tensor_dim0=1024, tile_dim0=1024, stride=1024}.
static __device__ __forceinline__ void tdm_load_8kb(unsigned lds_off, const void* gsrc) {
  unsigned long long ga = (unsigned long long)(size_t)gsrc;
  u32x4 g0;
  g0[0] = 1u;                                               // count=1 (valid user desc)
  g0[1] = lds_off;                                          // lds_addr (bytes)
  g0[2] = (unsigned)(ga & 0xFFFFFFFFu);                     // global_addr[31:0]
  g0[3] = (unsigned)((ga >> 32) & 0x01FFFFFFu) | (2u << 30);// global_addr[56:32] | type=2
  i32x8 g1;
  g1[0] = (3 << 16);          // workgroup_mask=0 (not in cluster), data_size=3 (8B)
  g1[1] = (1024 << 16);       // tensor_dim0[15:0] = 1024
  g1[2] = 0;                  // tensor_dim0 hi, tensor_dim1 lo
  g1[3] = (1024 << 16);       // tile_dim0 = 1024
  g1[4] = 0;                  // tile_dim1 = 0 (unused), tile_dim2 = 0
  g1[5] = 1024;               // tensor_dim0_stride = 1024
  g1[6] = 0;
  g1[7] = 0;
  i32x4 z4 = {0, 0, 0, 0};
#if __clang_major__ >= 23
  i32x8 z8 = {0, 0, 0, 0, 0, 0, 0, 0};
  __builtin_amdgcn_tensor_load_to_lds(g0, g1, z4, z4, z8, 0);
#else
  __builtin_amdgcn_tensor_load_to_lds(g0, g1, z4, z4, 0);
#endif
}
#endif

// ---------------- hash table ----------------

__global__ void init_hash_kernel(unsigned long long* keys, int cap) {
  int i = blockIdx.x * blockDim.x + threadIdx.x;
  if (i < cap) keys[i] = HEMPTY;
}

__global__ void hash_insert_kernel(const int* __restrict__ coords, int n,
                                   unsigned long long* __restrict__ keys,
                                   int* __restrict__ vals, int cap) {
  int i = blockIdx.x * blockDim.x + threadIdx.x;
  if (i >= n) return;
  unsigned long long key = lin3(coords[3 * i], coords[3 * i + 1], coords[3 * i + 2]);
  unsigned int mask = (unsigned)cap - 1u;
  unsigned int s = (unsigned int)hmix(key) & mask;
  for (;;) {
    unsigned long long prev = atomicCAS(&keys[s], HEMPTY, key);
    if (prev == HEMPTY || prev == key) { vals[s] = i; break; }
    s = (s + 1) & mask;
  }
}

__global__ void build_nb_kernel(const int* __restrict__ coords, int n, int stride,
                                const unsigned long long* __restrict__ keys,
                                const int* __restrict__ vals, int cap,
                                int* __restrict__ nb) {
  long long idx = (long long)blockIdx.x * blockDim.x + threadIdx.x;
  if (idx >= (long long)n * 27) return;
  int i = (int)(idx / 27), k = (int)(idx % 27);
  int dx = (k / 9) - 1, dy = ((k / 3) % 3) - 1, dz = (k % 3) - 1;
  unsigned long long key = lin3(coords[3 * i] + dx * stride,
                                coords[3 * i + 1] + dy * stride,
                                coords[3 * i + 2] + dz * stride);
  unsigned int mask = (unsigned)cap - 1u;
  unsigned int s = (unsigned int)hmix(key) & mask;
  int r = -1;
  for (int probe = 0; probe < cap; ++probe) {
    unsigned long long kk = keys[s];
    if (kk == key) { r = vals[s]; break; }
    if (kk == HEMPTY) break;
    s = (s + 1) & mask;
  }
  nb[(size_t)k * n + i] = r;
}

// ---------------- elementwise / packing ----------------

__global__ void requant_f16_kernel(const float* __restrict__ in, _Float16* __restrict__ out, long long cnt) {
  long long i = (long long)blockIdx.x * blockDim.x + threadIdx.x;
  if (i < cnt) out[i] = (_Float16)requantf(in[i]);
}

// pack conv weights [taps][64cin][64cout] f32 -> WMMA-B fragment order:
// [tap][kc(2)][nt(4)][lane(32)][e(16)] f16 (each lane's 16 halves contiguous)
__global__ void pack_wfrag_kernel(const float* __restrict__ W, _Float16* __restrict__ out, int taps) {
  long long idx = (long long)blockIdx.x * blockDim.x + threadIdx.x;
  if (idx >= (long long)taps * 4096) return;
  int k = (int)(idx >> 12), r = (int)(idx & 4095);
  int e = r & 15, lane = (r >> 4) & 31, nt = (r >> 9) & 3, kc = (r >> 11) & 1;
  int rlo = lane & 15, hi = lane >> 4;
  int kl = (e < 8) ? (hi * 8 + e) : (16 + hi * 8 + (e - 8));
  int cin = kc * 32 + kl, cout = nt * 16 + rlo;
  out[idx] = (_Float16)W[(size_t)k * 4096 + cin * 64 + cout];
}

// pack p1_Wl [64,255] f32 -> fragment order [nt(16)][kc(2)][lane(32)][e(16)] f16, col 255 padded 0
__global__ void pack_w255_kernel(const float* __restrict__ in, _Float16* __restrict__ out) {
  int idx = blockIdx.x * blockDim.x + threadIdx.x;  // 16*2*32*16 = 16384
  if (idx >= 16384) return;
  int e = idx & 15, lane = (idx >> 4) & 31, kc = (idx >> 9) & 1, nt = idx >> 10;
  int rlo = lane & 15, hi = lane >> 4;
  int kl = (e < 8) ? (hi * 8 + e) : (16 + hi * 8 + (e - 8));
  int cin = kc * 32 + kl, col = nt * 16 + rlo;
  out[idx] = (col < 255) ? (_Float16)in[cin * 255 + col] : (_Float16)0.0f;
}

// dec layer: h = prelu(requant([rec_F | 256*bins1]) @ W(72x64) + b, a)  (K=72, small; scalar)
__global__ void dec_kernel(const float* __restrict__ rec_F, const int* __restrict__ bins1,
                           const float* __restrict__ W, const float* __restrict__ b,
                           const float* __restrict__ a, float* __restrict__ h, int n) {
  long long idx = (long long)blockIdx.x * blockDim.x + threadIdx.x;
  if (idx >= (long long)n * 64) return;
  int i = (int)(idx >> 6), oc = (int)(idx & 63);
  float acc = b[oc];
  const float* xr = rec_F + (size_t)i * 64;
#pragma unroll 8
  for (int j = 0; j < 64; ++j) acc += requantf(xr[j]) * W[j * 64 + oc];
#pragma unroll
  for (int j = 0; j < 8; ++j)  // requant(bits*256) == bits
    acc += (float)bins1[i * 8 + j] * W[(64 + j) * 64 + oc];
  float av = a[oc];
  h[idx] = (acc >= 0.f) ? acc : av * acc;
}

// ---------------- sparse conv: 4 waves/block, TDM weight staging, WMMA f16 ----------------
__global__ __launch_bounds__(128) void sparse_conv_wmma_kernel(
    const int* __restrict__ nb, const _Float16* __restrict__ act,
    const _Float16* __restrict__ Wf,   // fragment-packed [27][8][512] f16
    const float* __restrict__ bias, const float* __restrict__ pa,
    const float* __restrict__ hadd, void* __restrict__ out, int out_is_f16, int n) {
  __shared__ __align__(16) _Float16 sW[4096];      // 8KB fragment-packed weight tile
  __shared__ __align__(16) _Float16 sA[4][1024];   // per-wave 16x64 gathered activations
  const int tid = threadIdx.x;
  const int wv = tid >> 5, lane = tid & 31;
  const int rlo = lane & 15, hi = lane >> 4;
  const int row0 = blockIdx.x * 64 + wv * 16;
#if HAVE_TDM
  const unsigned ldsW = (unsigned)(size_t)(void*)&sW[0];
#endif

  CFrag acc[4];
#pragma unroll
  for (int t = 0; t < 4; ++t)
#pragma unroll
    for (int r = 0; r < 8; ++r) acc[t].f[r] = 0.f;

  for (int k = 0; k < 27; ++k) {
    const _Float16* wsrc = Wf + (size_t)k * 4096;
#if HAVE_TDM
    if (wv == 0) {  // wave 0 drives the Tensor Data Mover for the shared weight tile
      tdm_load_8kb(ldsW, wsrc);
      __builtin_amdgcn_s_wait_tensorcnt((short)0);
    } else if (wv == 1 && k < 26) {
      __builtin_prefetch(Wf + (size_t)(k + 1) * 4096 + lane * 128, 0, 0);
    }
#else
    for (int t = tid; t < 512; t += 128) ((u32x4*)sW)[t] = ((const u32x4*)wsrc)[t];
#endif
    // gather this wave's 16 activation rows: lane -> row rlo, 32-col half hi
    {
      int grow = row0 + rlo;
      int src = (grow < n) ? nb[(size_t)k * n + grow] : -1;
      u32x4* adst = (u32x4*)(&sA[wv][rlo * 64 + hi * 32]);
      if (src >= 0) {
        const u32x4* asrc = (const u32x4*)(act + (size_t)src * 64 + hi * 32);
        adst[0] = asrc[0]; adst[1] = asrc[1]; adst[2] = asrc[2]; adst[3] = asrc[3];
      } else {
        u32x4 z = {0, 0, 0, 0};
        adst[0] = z; adst[1] = z; adst[2] = z; adst[3] = z;
      }
    }
    __syncthreads();
    // A fragments: ISA 16-bit A layout = two contiguous 8-half runs -> 2 x b128 LDS loads
    AFragQ afr[2];
#pragma unroll
    for (int kc = 0; kc < 2; ++kc) {
      const u32x4* ap = (const u32x4*)(&sA[wv][rlo * 64 + kc * 32 + hi * 8]);
      afr[kc].q[0] = ap[0];
      afr[kc].q[1] = ap[2];  // +16 halves
    }
#pragma unroll
    for (int nt = 0; nt < 4; ++nt) {
#pragma unroll
      for (int kc = 0; kc < 2; ++kc) {
        AFragQ bfr;
        const u32x4* bp = (const u32x4*)(sW + ((kc * 4 + nt) * 512 + lane * 16));
        bfr.q[0] = bp[0];
        bfr.q[1] = bp[1];
        acc[nt].v = __builtin_amdgcn_wmma_f32_16x16x32_f16(
            false, afr[kc].v, false, bfr.v, (short)0, acc[nt].v, false, false);
      }
    }
    __syncthreads();
  }
  // fused epilogue: C/D layout -> VGPR r holds M = r + 8*hi, N = rlo
#pragma unroll
  for (int nt = 0; nt < 4; ++nt) {
    int ch = nt * 16 + rlo;
    float bz = bias[ch];
#pragma unroll
    for (int r = 0; r < 8; ++r) {
      int grow = row0 + hi * 8 + r;
      if (grow >= n) continue;
      float v = acc[nt].f[r] + bz;
      if (out_is_f16) {
        float av = pa[ch];
        v = (v >= 0.f) ? v : av * v;
        ((_Float16*)out)[(size_t)grow * 64 + ch] = (_Float16)requantf(v);
      } else {
        v += hadd[(size_t)grow * 64 + ch];
        ((float*)out)[(size_t)grow * 64 + ch] = v;
      }
    }
  }
}

// child expansion: ch_q[m] = requant( pq[child>>3] @ p0_Wl[:, (child&7)*64 .. +64] )
// (B tile varies per row via child&7 -> not WMMA-tileable; ~3 GFLOP, scalar)
__global__ void child_gather_kernel(const _Float16* __restrict__ pq, const float* __restrict__ Wl,
                                    const int* __restrict__ child_idx, _Float16* __restrict__ out,
                                    int M) {
  long long idx = (long long)blockIdx.x * blockDim.x + threadIdx.x;
  if (idx >= (long long)M * 64) return;
  int m = (int)(idx >> 6), oc = (int)(idx & 63);
  int c = child_idx[m];
  int parent = c >> 3, b = c & 7;
  const _Float16* arow = pq + (size_t)parent * 64;
  const float* wcol = Wl + b * 64 + oc;
  float acc = 0.f;
#pragma unroll 8
  for (int j = 0; j < 64; ++j) acc += (float)arow[j] * wcol[(size_t)j * 512];
  out[idx] = (_Float16)requantf(acc);
}

// final linear 64 -> 255 (B fragment-packed+padded to 256) via WMMA; 4 waves/block
__global__ __launch_bounds__(128) void linear255_wmma_kernel(
    const _Float16* __restrict__ A, const _Float16* __restrict__ Bf,
    float* __restrict__ out, int M) {
  const int tid = threadIdx.x;
  const int wv = tid >> 5, lane = tid & 31;
  const int rlo = lane & 15, hi = lane >> 4;
  const int row0 = blockIdx.x * 64 + wv * 16;
  const int grow = row0 + rlo;
  const bool ok = grow < M;
  AFragQ afr[2];
#pragma unroll
  for (int kc = 0; kc < 2; ++kc) {
    u32x4 z = {0, 0, 0, 0};
    if (ok) {
      const u32x4* ap = (const u32x4*)(A + (size_t)grow * 64 + kc * 32 + hi * 8);
      afr[kc].q[0] = ap[0];
      afr[kc].q[1] = ap[2];  // +16 halves
    } else {
      afr[kc].q[0] = z;
      afr[kc].q[1] = z;
    }
  }
  for (int nt = 0; nt < 16; ++nt) {
    CFrag c;
#pragma unroll
    for (int r = 0; r < 8; ++r) c.f[r] = 0.f;
#pragma unroll
    for (int kc = 0; kc < 2; ++kc) {
      AFragQ bfr;
      const u32x4* bp = (const u32x4*)(Bf + ((nt * 2 + kc) * 512 + lane * 16));
      bfr.q[0] = bp[0];
      bfr.q[1] = bp[1];
      c.v = __builtin_amdgcn_wmma_f32_16x16x32_f16(
          false, afr[kc].v, false, bfr.v, (short)0, c.v, false, false);
    }
    int col = nt * 16 + rlo;
#pragma unroll
    for (int r = 0; r < 8; ++r) {
      int orow = row0 + hi * 8 + r;
      if (orow < M && col < 255) out[(size_t)orow * 255 + col] = c.f[r];
    }
  }
}

__global__ void curoct_kernel(const int* __restrict__ bins0, float* __restrict__ out, int M) {
  int m = blockIdx.x * blockDim.x + threadIdx.x;
  if (m >= M) return;
  int s = 0;
#pragma unroll
  for (int k = 0; k < 8; ++k) s += bins0[m * 8 + k] << k;
  out[m] = (float)(s - 1);
}

// ---------------- launch ----------------

extern "C" void kernel_launch(void* const* d_in, const int* in_sizes, int n_in,
                              void* d_out, int out_size, void* d_ws, size_t ws_size,
                              hipStream_t stream) {
  const float* rec_F  = (const float*)d_in[0];
  const float* dec_W  = (const float*)d_in[1];
  const float* dec_b  = (const float*)d_in[2];
  const float* dec_a  = (const float*)d_in[3];
  const float* res_W1 = (const float*)d_in[4];
  const float* res_b1 = (const float*)d_in[5];
  const float* res_a1 = (const float*)d_in[6];
  const float* res_W2 = (const float*)d_in[7];
  const float* res_b2 = (const float*)d_in[8];
  const float* p0_Wc  = (const float*)d_in[9];
  const float* p0_bc  = (const float*)d_in[10];
  const float* p0_ac  = (const float*)d_in[11];
  const float* p0_Wl  = (const float*)d_in[12];
  const float* p1_Wc  = (const float*)d_in[13];
  const float* p1_bc  = (const float*)d_in[14];
  const float* p1_ac  = (const float*)d_in[15];
  const float* p1_Wl  = (const float*)d_in[16];
  const int* rec_C     = (const int*)d_in[17];
  const int* bins1     = (const int*)d_in[18];
  const int* child_idx = (const int*)d_in[19];
  const int* child_C   = (const int*)d_in[20];
  const int* bins0     = (const int*)d_in[21];
  (void)n_in; (void)out_size; (void)ws_size;

  const int N = in_sizes[17] / 3;
  const int M = in_sizes[19];

  char* p = (char*)d_ws;
  auto carve = [&](size_t bytes) -> char* {
    char* r = p;
    p += (bytes + 255) & ~(size_t)255;
    return r;
  };
  const int RC = 1 << 19, CC = 1 << 20;
  unsigned long long* recK = (unsigned long long*)carve((size_t)RC * 8);
  int* recV = (int*)carve((size_t)RC * 4);
  unsigned long long* chK = (unsigned long long*)carve((size_t)CC * 8);
  int* chV = (int*)carve((size_t)CC * 4);
  int* nbR = (int*)carve((size_t)27 * N * 4);
  int* nbC = (int*)carve((size_t)27 * M * 4);
  float* hbuf = (float*)carve((size_t)N * 64 * 4);
  size_t mxrows = (size_t)(N > M ? N : M);
  _Float16* bufA = (_Float16*)carve(mxrows * 64 * 2);
  _Float16* bufB = (_Float16*)carve(mxrows * 64 * 2);
  _Float16* w1f  = (_Float16*)carve((size_t)27 * 4096 * 2);
  _Float16* w2f  = (_Float16*)carve((size_t)27 * 4096 * 2);
  _Float16* w0cf = (_Float16*)carve((size_t)27 * 4096 * 2);
  _Float16* w1cf = (_Float16*)carve((size_t)27 * 4096 * 2);
  _Float16* wp1f = (_Float16*)carve((size_t)16384 * 2);

  float* out_rec  = (float*)d_out;
  float* out_pred = out_rec + (size_t)N * 64;
  float* out_oct  = out_pred + (size_t)M * 255;

  const int TB = 256;
  auto g1 = [&](long long cnt) { return (unsigned)((cnt + TB - 1) / TB); };

  // 1) hash tables + neighbor lists (reused by every conv on that coordinate set)
  init_hash_kernel<<<g1(RC), TB, 0, stream>>>(recK, RC);
  init_hash_kernel<<<g1(CC), TB, 0, stream>>>(chK, CC);
  hash_insert_kernel<<<g1(N), TB, 0, stream>>>(rec_C, N, recK, recV, RC);
  hash_insert_kernel<<<g1(M), TB, 0, stream>>>(child_C, M, chK, chV, CC);
  build_nb_kernel<<<g1((long long)N * 27), TB, 0, stream>>>(rec_C, N, 2, recK, recV, RC, nbR);
  build_nb_kernel<<<g1((long long)M * 27), TB, 0, stream>>>(child_C, M, 1, chK, chV, CC, nbC);

  // 2) weight packing into WMMA fragment order
  pack_wfrag_kernel<<<g1(27LL * 4096), TB, 0, stream>>>(res_W1, w1f, 27);
  pack_wfrag_kernel<<<g1(27LL * 4096), TB, 0, stream>>>(res_W2, w2f, 27);
  pack_wfrag_kernel<<<g1(27LL * 4096), TB, 0, stream>>>(p0_Wc, w0cf, 27);
  pack_wfrag_kernel<<<g1(27LL * 4096), TB, 0, stream>>>(p1_Wc, w1cf, 27);
  pack_w255_kernel<<<g1(16384), TB, 0, stream>>>(p1_Wl, wp1f);

  // 3) dec: h = prelu(requant(x) @ dec_W + b)
  dec_kernel<<<g1((long long)N * 64), TB, 0, stream>>>(rec_F, bins1, dec_W, dec_b, dec_a, hbuf, N);

  const unsigned gridN = (unsigned)((N + 63) / 64);
  const unsigned gridM = (unsigned)((M + 63) / 64);

  // 4) ResBlock: conv1 (prelu+requant fused) -> conv2 (+h residual, f32 -> d_out)
  requant_f16_kernel<<<g1((long long)N * 64), TB, 0, stream>>>(hbuf, bufA, (long long)N * 64);
  sparse_conv_wmma_kernel<<<gridN, 128, 0, stream>>>(nbR, bufA, w1f, res_b1, res_a1, nullptr, bufB, 1, N);
  sparse_conv_wmma_kernel<<<gridN, 128, 0, stream>>>(nbR, bufB, w2f, res_b2, nullptr, hbuf, out_rec, 0, N);

  // 5) pred[0]: requant(rec_out) -> conv (prelu+requant fused) -> child gather-GEMM (requant fused)
  requant_f16_kernel<<<g1((long long)N * 64), TB, 0, stream>>>(out_rec, bufA, (long long)N * 64);
  sparse_conv_wmma_kernel<<<gridN, 128, 0, stream>>>(nbR, bufA, w0cf, p0_bc, p0_ac, nullptr, bufB, 1, N);
  child_gather_kernel<<<g1((long long)M * 64), TB, 0, stream>>>(bufB, p0_Wl, child_idx, bufA, M);

  // 6) pred[1]: conv on child coords (prelu+requant fused) -> 64x255 WMMA GEMM
  sparse_conv_wmma_kernel<<<gridM, 128, 0, stream>>>(nbC, bufA, w1cf, p1_bc, p1_ac, nullptr, bufB, 1, M);
  linear255_wmma_kernel<<<gridM, 128, 0, stream>>>(bufB, wp1f, out_pred, M);

  // 7) cur_oct
  curoct_kernel<<<g1(M), TB, 0, stream>>>(bins0, out_oct, M);
}
